// MultiqueryScaledDotProductAttention_37374805410020
// MI455X (gfx1250) — matrix operand order repaired and checked by reading, not compiled, not run on hardware
//
#include <hip/hip_runtime.h>

typedef __attribute__((ext_vector_type(16))) _Float16 v16h;
typedef __attribute__((ext_vector_type(8)))  float    v8f;
typedef __attribute__((ext_vector_type(4)))  unsigned int u32x4;
typedef __attribute__((ext_vector_type(8)))  int      i32x8;
typedef __attribute__((ext_vector_type(4)))  int      i32x4;

#define S_  2048
#define E_  1024
#define H_  16
#define HD  64
#define NEGBIG (-1.0e30f)

// ---------- workspace layout (in halves, all 16B aligned) ----------
#define OFF_XH  ((size_t)0)          // [S, E]
#define OFF_WQ  ((size_t)2097152)    // [E, E]
#define OFF_WK  ((size_t)3145728)    // [HD, E]
#define OFF_WV  ((size_t)3211264)    // [HD, E]
#define OFF_WO  ((size_t)3276800)    // [E, E]
#define OFF_QH  ((size_t)4325376)    // [S, E]   (== [S,H,HD]), pre-scaled by 1/8
#define OFF_KH  ((size_t)6422528)    // [S, HD]
#define OFF_VT  ((size_t)6553600)    // [HD, S]  (V transposed)
#define OFF_AH  ((size_t)6684672)    // [S, E]   attention output (pre out-proj)

#if defined(__has_builtin)
#if __has_builtin(__builtin_amdgcn_tensor_load_to_lds)
#define HAVE_TDM 1
#endif
#endif

__device__ __forceinline__ v8f wmma16(v16h a, v16h b, v8f c) {
  return __builtin_amdgcn_wmma_f32_16x16x32_f16(
      /*neg_a=*/false, a, /*neg_b=*/false, b,
      /*c_mod=*/(short)0, c, /*reuse_a=*/false, /*reuse_b=*/false);
}

// A-matrix 16x32 f16 fragment (ISA 7.12.2): lane = row; VGPR d, half s ->
// K = (d&3)*2 + s + 8*(lane>=16) + 16*(d>=4)
__device__ __forceinline__ v16h load_fragA(const _Float16* row, int hi) {
  v16h f;
#pragma unroll
  for (int d = 0; d < 8; ++d) {
    const int k0 = ((d & 3) << 1) + (hi << 3) + ((d >> 2) << 4);
    f[2 * d]     = row[k0];
    f[2 * d + 1] = row[k0 + 1];
  }
  return f;
}

// B-matrix 32x16 f16 fragment: lane = col; element i ->
// K = 8*hi + i (i<8), 16 + 8*hi + (i-8) (i>=8) (same order as fragA)
__device__ __forceinline__ v16h load_fragB(const _Float16* row, int hi) {
  v16h f;
  const _Float16* p = row + (hi << 3);
#pragma unroll
  for (int i = 0; i < 16; ++i) f[i] = p[i + ((i >> 3) << 3)];
  return f;
}

#ifdef HAVE_TDM
// Issue one TDM 2D tile load: tile_x (contiguous, elems) x tile_y rows,
// row stride = stride_elems, f16 elements, packed row-major into LDS.
// Tensor dims == tile dims (tiles never go OOB here).
__device__ __forceinline__ void tdm_load_2d(unsigned lds_off, const _Float16* gptr,
                                            unsigned tile_x, unsigned tile_y,
                                            unsigned stride_elems) {
  unsigned long long ga = (unsigned long long)(uintptr_t)gptr;
  u32x4 g0 = {};
  g0[0] = 1u;                                        // count=1, no gather, user
  g0[1] = lds_off;                                   // LDS byte address
  g0[2] = (unsigned)(ga & 0xffffffffu);              // global_addr[31:0]
  g0[3] = (unsigned)((ga >> 32) & 0x01ffffffu)       // global_addr[56:32]
        | (2u << 30);                                // type = 2 ("image")
  i32x8 g1 = {};
  g1[0] = (int)(1u << 16);                           // data_size = 1 -> 2 bytes
  g1[1] = (int)((tile_x & 0xffffu) << 16);           // tensor_dim0[15:0]
  g1[2] = (int)(((tile_x >> 16) & 0xffffu)           // tensor_dim0[31:16]
        | ((tile_y & 0xffffu) << 16));               // tensor_dim1[15:0]
  g1[3] = (int)(((tile_y >> 16) & 0xffffu)           // tensor_dim1[31:16]
        | ((tile_x & 0xffffu) << 16));               // tile_dim0
  g1[4] = (int)(tile_y & 0xffffu);                   // tile_dim1 (tile_dim2 = 0)
  g1[5] = (int)stride_elems;                         // tensor_dim0_stride[31:0]
  g1[6] = 0;                                         // stride hi, dim1_stride
  g1[7] = 0;
  i32x4 z4 = {};
#if __clang_major__ >= 23
  i32x8 z8 = {};
  __builtin_amdgcn_tensor_load_to_lds(g0, g1, z4, z4, z8, 0);
#else
  __builtin_amdgcn_tensor_load_to_lds(g0, g1, z4, z4, 0);
#endif
}
#endif

// ---------------- f32 -> f16 conversion ----------------
__global__ __launch_bounds__(256) void cvt_f32_f16(const float* __restrict__ src,
                                                   _Float16* __restrict__ dst, int n) {
  int i = blockIdx.x * 256 + threadIdx.x;
  if (i < n) dst[i] = (_Float16)src[i];
}

// ---------------- generic WMMA GEMM: C[M,N] = A[M,K] @ W[N,K]^T + bias ----------------
// grid = (M/128, N/64), block = 256 (8 waves). Wave w owns rows w*16..w*16+15, 4 N-tiles.
// Double-buffered TDM staging: DMA of tile kt+1 overlaps WMMA of tile kt.
// MODE: 0 = f16 row-major out, 1 = f16 transposed out, 2 = f32 row-major out
template <int MODE>
__global__ __launch_bounds__(256) void gemm_xwt(
    const _Float16* __restrict__ A, const _Float16* __restrict__ W,
    const float* __restrict__ bias, void* __restrict__ outp,
    int Kd, int ldo, float scale) {
  __shared__ __align__(16) _Float16 As[2 * 128 * 32];
  __shared__ __align__(16) _Float16 Bs[2 * 64 * 32];

  const int tid  = threadIdx.x;
  const int wid  = tid >> 5;
  const int lane = tid & 31;
  const int hi   = lane >> 4;
  const int ln   = lane & 15;
  const int mBase = blockIdx.x * 128;
  const int nBase = blockIdx.y * 64;

  v8f acc[4];
  v8f zero = {};
#pragma unroll
  for (int d = 0; d < 4; ++d) acc[d] = zero;

  const int nkt = Kd >> 5;
#ifdef HAVE_TDM
  const bool issuer = (__builtin_amdgcn_readfirstlane(wid) == 0);
  if (issuer) {  // prime buffer 0
    tdm_load_2d((unsigned)(uintptr_t)&As[0], A + (size_t)mBase * Kd, 32u, 128u,
                (unsigned)Kd);
    tdm_load_2d((unsigned)(uintptr_t)&Bs[0], W + (size_t)nBase * Kd, 32u, 64u,
                (unsigned)Kd);
  }
#endif
  for (int kt = 0; kt < nkt; ++kt) {
    _Float16* Ab = As + (kt & 1) * (128 * 32);
    _Float16* Bb = Bs + (kt & 1) * (64 * 32);
#ifdef HAVE_TDM
    if (issuer) {
      if (kt + 1 < nkt) {  // kick DMA for next tile into the other buffer
        const int kOff2 = (kt + 1) << 5;
        const int nb = (kt + 1) & 1;
        tdm_load_2d((unsigned)(uintptr_t)&As[nb * (128 * 32)],
                    A + (size_t)mBase * Kd + kOff2, 32u, 128u, (unsigned)Kd);
        tdm_load_2d((unsigned)(uintptr_t)&Bs[nb * (64 * 32)],
                    W + (size_t)nBase * Kd + kOff2, 32u, 64u, (unsigned)Kd);
        // 2 newest ops may stay outstanding; TDM completes in order,
        // so <=2 outstanding means tile kt has fully landed.
#if __has_builtin(__builtin_amdgcn_s_wait_tensorcnt)
        __builtin_amdgcn_s_wait_tensorcnt((short)2);
#else
        __asm__ volatile("s_wait_tensorcnt 0x2" ::: "memory");
#endif
      } else {
#if __has_builtin(__builtin_amdgcn_s_wait_tensorcnt)
        __builtin_amdgcn_s_wait_tensorcnt((short)0);
#else
        __asm__ volatile("s_wait_tensorcnt 0x0" ::: "memory");
#endif
      }
    }
#else
    // fallback: VGPR-bounce staging
    const int kOff = kt << 5;
#pragma unroll
    for (int it = 0; it < 2; ++it) {
      int li  = tid + it * 256;
      int row = li >> 2, c4 = li & 3;
      *(uint4*)(Ab + row * 32 + c4 * 8) =
          *(const uint4*)(A + (size_t)(mBase + row) * Kd + kOff + c4 * 8);
    }
    {
      int row = tid >> 2, c4 = tid & 3;
      *(uint4*)(Bb + row * 32 + c4 * 8) =
          *(const uint4*)(W + (size_t)(nBase + row) * Kd + kOff + c4 * 8);
    }
#endif
    __syncthreads();

    v16h af = load_fragA(Ab + (wid * 16 + ln) * 32, hi);
#pragma unroll
    for (int d = 0; d < 4; ++d) {
      v16h bf = load_fragB(Bb + (d * 16 + ln) * 32, hi);
      acc[d] = wmma16(af, bf, acc[d]);
    }
    __syncthreads();
  }

  // epilogue: C-frag lane L, VGPR j -> M = j + 8*hi, N = ln
#pragma unroll
  for (int d = 0; d < 4; ++d) {
#pragma unroll
    for (int j = 0; j < 8; ++j) {
      const int row = mBase + wid * 16 + j + 8 * hi;
      const int col = nBase + d * 16 + ln;
      const float v = (acc[d][j] + bias[col]) * scale;
      if (MODE == 2) {
        ((float*)outp)[(size_t)row * ldo + col] = v;
      } else if (MODE == 1) {
        ((_Float16*)outp)[(size_t)col * ldo + row] = (_Float16)v;
      } else {
        ((_Float16*)outp)[(size_t)row * ldo + col] = (_Float16)v;
      }
    }
  }
}

// ---------------- flash attention: one wave per (head, 16-query tile) ----------------
// grid = H*(S/16)/8 = 256 blocks, block = 256 threads (8 waves)
__global__ __launch_bounds__(256) void attn_flash(
    const _Float16* __restrict__ Qh, const _Float16* __restrict__ Kh,
    const _Float16* __restrict__ Vt, _Float16* __restrict__ Ah) {
  __shared__ __align__(16) _Float16 Pl[8][16 * 32];

  const int tid  = threadIdx.x;
  const int wid  = tid >> 5;
  const int lane = tid & 31;
  const int hi   = lane >> 4;
  const int ln   = lane & 15;
  // h/qt are wave-uniform: make that visible to the compiler so all loop /
  // mask control flow is scalar and WMMA always executes with full EXEC.
  const int gw   = __builtin_amdgcn_readfirstlane(blockIdx.x * 8 + wid);
  const int h    = gw >> 7;      // head
  const int qt   = gw & 127;     // query tile
  const int qBase = qt * 16;

  // Q fragments (already scaled by 1/sqrt(D) at projection time)
  const _Float16* qrow = Qh + (size_t)(qBase + ln) * E_ + h * HD;
  const v16h qa0 = load_fragA(qrow, hi);
  const v16h qa1 = load_fragA(qrow + 32, hi);

  float m[8], l[8];
  v8f acc[4];
  v8f zero = {};
#pragma unroll
  for (int j = 0; j < 8; ++j) { m[j] = NEGBIG; l[j] = 0.0f; }
#pragma unroll
  for (int d = 0; d < 4; ++d) acc[d] = zero;

  const int nk = (qt >> 1) + 1;  // 32-key chunks covering keys 0..qBase+15
  for (int c = 0; c < nk; ++c) {
    const int kb = c * 32;
    if (c + 1 < nk)  // near-scope prefetch of next K chunk
      __builtin_prefetch(Kh + (size_t)(kb + 32 + ln) * HD, 0, 3);

    // ---- scores: two 16-key tiles, each Q(16x64) @ K^T(64x16), K split 2x32
    v8f s0 = zero, s1;
    {
      const _Float16* kr = Kh + (size_t)(kb + ln) * HD;
      v16h b0 = load_fragB(kr, hi);
      v16h b1 = load_fragB(kr + 32, hi);
      s0 = wmma16(qa0, b0, s0);
      s0 = wmma16(qa1, b1, s0);
      if (kb == qBase) {  // diagonal tile -> causal mask
#pragma unroll
        for (int j = 0; j < 8; ++j)
          s0[j] = (kb + ln > qBase + j + 8 * hi) ? NEGBIG : s0[j];
      }
    }
    const int ktb1 = kb + 16;
    if (ktb1 <= qBase) {
      v8f t = zero;
      const _Float16* kr = Kh + (size_t)(ktb1 + ln) * HD;
      v16h b0 = load_fragB(kr, hi);
      v16h b1 = load_fragB(kr + 32, hi);
      t = wmma16(qa0, b0, t);
      t = wmma16(qa1, b1, t);
      if (ktb1 == qBase) {
#pragma unroll
        for (int j = 0; j < 8; ++j)
          t[j] = (ktb1 + ln > qBase + j + 8 * hi) ? NEGBIG : t[j];
      }
      s1 = t;
    } else {
#pragma unroll
      for (int j = 0; j < 8; ++j) s1[j] = NEGBIG;
    }

    // ---- online softmax update (rows live across 16-lane groups)
#pragma unroll
    for (int j = 0; j < 8; ++j) {
      float cm = fmaxf(s0[j], s1[j]);
      cm = fmaxf(cm, __shfl_xor(cm, 1, 32));
      cm = fmaxf(cm, __shfl_xor(cm, 2, 32));
      cm = fmaxf(cm, __shfl_xor(cm, 4, 32));
      cm = fmaxf(cm, __shfl_xor(cm, 8, 32));
      const float mn    = fmaxf(m[j], cm);
      const float alpha = __expf(m[j] - mn);
      const float p0 = __expf(s0[j] - mn);
      const float p1 = __expf(s1[j] - mn);
      float rs = p0 + p1;
      rs += __shfl_xor(rs, 1, 32);
      rs += __shfl_xor(rs, 2, 32);
      rs += __shfl_xor(rs, 4, 32);
      rs += __shfl_xor(rs, 8, 32);
      l[j] = l[j] * alpha + rs;
      m[j] = mn;
#pragma unroll
      for (int d = 0; d < 4; ++d) acc[d][j] *= alpha;
      const int pr = j + 8 * hi;
      Pl[wid][pr * 32 + ln]      = (_Float16)p0;
      Pl[wid][pr * 32 + 16 + ln] = (_Float16)p1;
    }

    // in-wave LDS turnaround: stores above -> transposed fragment reads below
    __asm__ volatile("s_wait_dscnt 0" ::: "memory");

    // ---- O += P(16x32) @ V(32x16 per dim-tile)
    v16h pa = load_fragA(&Pl[wid][ln * 32], hi);
#pragma unroll
    for (int d = 0; d < 4; ++d) {
      v16h bv = load_fragB(Vt + (size_t)(d * 16 + ln) * S_ + kb, hi);
      acc[d] = wmma16(pa, bv, acc[d]);
    }
  }

  // ---- finalize: O /= l, store f16 into Ah[S, E] at column h*64
#pragma unroll
  for (int j = 0; j < 8; ++j) {
    const float inv = 1.0f / l[j];
    const int srow = qBase + j + 8 * hi;
#pragma unroll
    for (int d = 0; d < 4; ++d)
      Ah[(size_t)srow * E_ + h * HD + d * 16 + ln] = (_Float16)(acc[d][j] * inv);
  }
}

extern "C" void kernel_launch(void* const* d_in, const int* in_sizes, int n_in,
                              void* d_out, int out_size, void* d_ws, size_t ws_size,
                              hipStream_t stream) {
  (void)in_sizes; (void)n_in; (void)out_size; (void)ws_size;
  const float* X  = (const float*)d_in[0];
  // d_in[1] = mask (causal, known analytically -> ignored)
  const float* Wq = (const float*)d_in[2];
  const float* bq = (const float*)d_in[3];
  const float* Wk = (const float*)d_in[4];
  const float* bk = (const float*)d_in[5];
  const float* Wv = (const float*)d_in[6];
  const float* bv = (const float*)d_in[7];
  const float* Wo = (const float*)d_in[8];
  const float* bo = (const float*)d_in[9];
  float* out = (float*)d_out;

  _Float16* ws  = (_Float16*)d_ws;
  _Float16* Xh  = ws + OFF_XH;
  _Float16* Wqh = ws + OFF_WQ;
  _Float16* Wkh = ws + OFF_WK;
  _Float16* Wvh = ws + OFF_WV;
  _Float16* Woh = ws + OFF_WO;
  _Float16* Qh  = ws + OFF_QH;
  _Float16* Kh  = ws + OFF_KH;
  _Float16* Vt  = ws + OFF_VT;
  _Float16* Ah  = ws + OFF_AH;

  // 1) f32 -> f16 conversions
  cvt_f32_f16<<<(S_ * E_ + 255) / 256, 256, 0, stream>>>(X, Xh, S_ * E_);
  cvt_f32_f16<<<(E_ * E_ + 255) / 256, 256, 0, stream>>>(Wq, Wqh, E_ * E_);
  cvt_f32_f16<<<(HD * E_ + 255) / 256, 256, 0, stream>>>(Wk, Wkh, HD * E_);
  cvt_f32_f16<<<(HD * E_ + 255) / 256, 256, 0, stream>>>(Wv, Wvh, HD * E_);
  cvt_f32_f16<<<(E_ * E_ + 255) / 256, 256, 0, stream>>>(Wo, Woh, E_ * E_);

  // 2) projections (Q scaled by 1/sqrt(64) = 0.125 here; V stored transposed)
  gemm_xwt<0><<<dim3(S_ / 128, E_ / 64), 256, 0, stream>>>(
      Xh, Wqh, bq, (void*)Qh, E_, E_, 0.125f);
  gemm_xwt<0><<<dim3(S_ / 128, 1), 256, 0, stream>>>(
      Xh, Wkh, bk, (void*)Kh, E_, HD, 1.0f);
  gemm_xwt<1><<<dim3(S_ / 128, 1), 256, 0, stream>>>(
      Xh, Wvh, bv, (void*)Vt, E_, S_, 1.0f);

  // 3) flash attention: H * (S/16) wave-tiles, 8 waves/block
  attn_flash<<<(H_ * (S_ / 16)) / 8, 256, 0, stream>>>(Qh, Kh, Vt, Ah);

  // 4) output projection, f32 result + bias straight to d_out
  gemm_xwt<2><<<dim3(S_ / 128, E_ / 64), 256, 0, stream>>>(
      Ah, Woh, bo, (void*)out, E_, E_, 1.0f);
}
